// MMDLoss_77438260347128
// MI455X (gfx1250) — matrix operand order repaired and checked by reading, not compiled, not run on hardware
//
#include <hip/hip_runtime.h>

// ---------------------------------------------------------------------------
// MMD loss (Gaussian kernel, sigma=1) for feat_s/feat_t = f32[8192,256].
// Compute-bound: 3x (8192x8192x256) GEMMs -> v_wmma_f32_16x16x32_bf16 with
// hi/lo bf16 splitting (~f32-quality dots). k_ss / k_tt are symmetric, so
// upper-triangle block tiles are skipped and lower ones doubled (-33% work).
// ---------------------------------------------------------------------------

typedef __attribute__((ext_vector_type(16))) __bf16 v16bf;
typedef __attribute__((ext_vector_type(8)))  float  v8f;

constexpr int N  = 8192;   // rows in each feature matrix
constexpr int D  = 256;    // feature dim (K)
constexpr int BT = 128;    // square block tile (rows == cols, transpose-symmetric)
constexpr int BK = 32;     // K step (matches 16x16x32 WMMA)
constexpr int LDA = 40;    // LDS A row pitch in bf16 units (80B, 16B aligned)
constexpr int LDB = 136;   // LDS B row pitch in bf16 units (272B, 16B aligned)
constexpr int NT = 512;    // threads per tile block (16 waves, 4x4 of 32x32)

union FragA { v16bf v; unsigned u[8]; };
union FragB { v16bf v; uint4 q[2]; };
union Acc   { v8f  v; float f[8]; };

__device__ __forceinline__ unsigned short f32_to_bf16_rne(float f) {
  unsigned u = __float_as_uint(f);
  u += 0x7FFFu + ((u >> 16) & 1u);   // round-to-nearest-even (data is finite)
  return (unsigned short)(u >> 16);
}
__device__ __forceinline__ float bf16_to_f32(unsigned short h) {
  return __uint_as_float(((unsigned)h) << 16);
}

// ---------------------------------------------------------------------------
// Kernel 1: split f32 features into bf16 hi/lo and compute row norms (f32).
// grid = (N, 2), block = D (=256) threads; blockIdx.y selects s (0) / t (1).
// ---------------------------------------------------------------------------
__global__ __launch_bounds__(256) void mmd_prep(
    const float* __restrict__ fs, const float* __restrict__ ft,
    unsigned short* __restrict__ s_hi, unsigned short* __restrict__ s_lo,
    unsigned short* __restrict__ t_hi, unsigned short* __restrict__ t_lo,
    float* __restrict__ x2s, float* __restrict__ x2t) {
  const int row = blockIdx.x;
  const int z   = blockIdx.y;
  const float* src = z ? ft : fs;
  unsigned short* hi = z ? t_hi : s_hi;
  unsigned short* lo = z ? t_lo : s_lo;
  float* nrm = z ? x2t : x2s;

  const int t = threadIdx.x;
  const float v = src[(size_t)row * D + t];
  const unsigned short h = f32_to_bf16_rne(v);
  const unsigned short l = f32_to_bf16_rne(v - bf16_to_f32(h));
  hi[(size_t)row * D + t] = h;
  lo[(size_t)row * D + t] = l;

  __shared__ float red[256];
  red[t] = v * v;
  __syncthreads();
  #pragma unroll
  for (int s = 128; s > 0; s >>= 1) {
    if (t < s) red[t] += red[t + s];
    __syncthreads();
  }
  if (t == 0) nrm[row] = red[0];
}

// ---------------------------------------------------------------------------
// Kernel 2: tiled WMMA GEMM fused with gaussian-kernel epilogue + block sum.
// grid = (N/BT, N/BT, 3); z: 0=ss, 1=tt (symmetric: lower blocks x2, upper
// blocks skipped), 2=st (full, weight -2).
// Block = 512 threads = 16 waves in a 4x4 arrangement; wave tile = 32x32.
// ---------------------------------------------------------------------------
__global__ __launch_bounds__(NT) void mmd_tiles(
    const unsigned short* __restrict__ s_hi, const unsigned short* __restrict__ s_lo,
    const unsigned short* __restrict__ t_hi, const unsigned short* __restrict__ t_lo,
    const float* __restrict__ x2s, const float* __restrict__ x2t,
    float* __restrict__ partials) {
  __shared__ unsigned short As[2][BT * LDA];   // [hi/lo][m][k], pitch LDA
  __shared__ unsigned short Bs[2][BK * LDB];   // [hi/lo][k][n], pitch LDB
  __shared__ float xs[BT];
  __shared__ float ys[BT];
  __shared__ float red[NT];

  const int z  = blockIdx.z;
  const int bm = blockIdx.y * BT;
  const int bn = blockIdx.x * BT;
  const int t  = threadIdx.x;
  const int pidx = (z * gridDim.y + blockIdx.y) * gridDim.x + blockIdx.x;

  // Symmetry: for ss/tt only compute lower-triangle block tiles.
  if (z < 2 && bn > bm) {
    if (t == 0) partials[pidx] = 0.0f;
    return;
  }
  // Weight: lower off-diagonal blocks stand in for their mirror too.
  const float w = (z == 2) ? -2.0f : ((bn == bm) ? 1.0f : 2.0f);

  const unsigned short* Ahi = (z == 1) ? t_hi : s_hi;
  const unsigned short* Alo = (z == 1) ? t_lo : s_lo;
  const unsigned short* Bhi = (z == 0) ? s_hi : t_hi;
  const unsigned short* Blo = (z == 0) ? s_lo : t_lo;
  const float* xn = (z == 1) ? x2t : x2s;
  const float* yn = (z == 0) ? x2s : x2t;

  if (t < BT) {
    xs[t] = xn[bm + t];
    ys[t] = yn[bn + t];
  }

  const int lane  = t & 31;
  const int wid   = t >> 5;
  const int wm    = (wid >> 2) * 32;   // wave row offset: 0,32,64,96
  const int wn    = (wid & 3)  * 32;   // wave col offset: 0,32,64,96
  const int row16 = lane & 15;
  const int half  = lane >> 4;

  Acc acc[2][2];
  #pragma unroll
  for (int mt = 0; mt < 2; ++mt)
    #pragma unroll
    for (int nt = 0; nt < 2; ++nt)
      #pragma unroll
      for (int r = 0; r < 8; ++r) acc[mt][nt].f[r] = 0.0f;

  // A-tile copy: each thread copies 8 bf16 (1x b128) per buffer.
  const int am = t >> 2;              // 0..127 (m)
  const int ap = (t & 3) * 8;         // 0,8,16,24 (k)
  // B-tile copy (transpose): each thread loads 8 bf16 of a y-row, scatters.
  const int btn = t >> 2;             // 0..127 (n)
  const int bkq = (t & 3) * 8;        // 0,8,16,24 (k)

  for (int kb = 0; kb < D; kb += BK) {
    __syncthreads();   // previous iter's LDS reads complete

    // --- stage A (row-major) ---
    {
      *(uint4*)(&As[0][am * LDA + ap]) =
          *(const uint4*)(Ahi + (size_t)(bm + am) * D + kb + ap);
      *(uint4*)(&As[1][am * LDA + ap]) =
          *(const uint4*)(Alo + (size_t)(bm + am) * D + kb + ap);
    }
    // --- stage B transposed: global [n][k] -> LDS [k][n] ---
    {
      uint4 vh = *(const uint4*)(Bhi + (size_t)(bn + btn) * D + kb + bkq);
      uint4 vl = *(const uint4*)(Blo + (size_t)(bn + btn) * D + kb + bkq);
      const unsigned short* ph = (const unsigned short*)&vh;
      const unsigned short* pl = (const unsigned short*)&vl;
      #pragma unroll
      for (int i = 0; i < 8; ++i) {
        Bs[0][(bkq + i) * LDB + btn] = ph[i];
        Bs[1][(bkq + i) * LDB + btn] = pl[i];
      }
    }
    __syncthreads();

    // --- build A fragments (ISA 16-bit A 16x32 layout: lane=M, K split by half) ---
    FragA ah[2], al[2];
    #pragma unroll
    for (int mt = 0; mt < 2; ++mt) {
      const int mrow = wm + mt * 16 + row16;
      const unsigned short* bh = &As[0][mrow * LDA];
      const unsigned short* bl = &As[1][mrow * LDA];
      #pragma unroll
      for (int j = 0; j < 4; ++j) {
        const int k0 = half * 8 + 2 * j;
        ah[mt].u[j]     = *(const unsigned*)(bh + k0);
        ah[mt].u[4 + j] = *(const unsigned*)(bh + 16 + k0);
        al[mt].u[j]     = *(const unsigned*)(bl + k0);
        al[mt].u[4 + j] = *(const unsigned*)(bl + 16 + k0);
      }
    }
    // --- build B fragments (lane = N, elements = K in same split as A) ---
    FragB bh[2], bl[2];
    #pragma unroll
    for (int nt = 0; nt < 2; ++nt) {
      const int ncol = wn + nt * 16 + row16;
      unsigned short* eh = (unsigned short*)&bh[nt];
      unsigned short* el = (unsigned short*)&bl[nt];
      #pragma unroll
      for (int s = 0; s < 16; ++s) {
        const int kk = (s < 8) ? (half * 8 + s) : (16 + half * 8 + (s - 8));
        eh[s] = Bs[0][kk * LDB + ncol];
        el[s] = Bs[1][kk * LDB + ncol];
      }
    }

    // --- 12 WMMAs: hi*hi + hi*lo + lo*hi (drop lo*lo) ---
    #pragma unroll
    for (int mt = 0; mt < 2; ++mt)
      #pragma unroll
      for (int nt = 0; nt < 2; ++nt) {
        acc[mt][nt].v = __builtin_amdgcn_wmma_f32_16x16x32_bf16(
            false, ah[mt].v, false, bh[nt].v, (short)0, acc[mt][nt].v, false, false);
        acc[mt][nt].v = __builtin_amdgcn_wmma_f32_16x16x32_bf16(
            false, ah[mt].v, false, bl[nt].v, (short)0, acc[mt][nt].v, false, false);
        acc[mt][nt].v = __builtin_amdgcn_wmma_f32_16x16x32_bf16(
            false, al[mt].v, false, bh[nt].v, (short)0, acc[mt][nt].v, false, false);
      }
  }

  // --- fused epilogue: d = x2 + y2 - 2xy; sum exp(-d/2) (weighted) ---
  float lsum = 0.0f;
  #pragma unroll
  for (int mt = 0; mt < 2; ++mt)
    #pragma unroll
    for (int nt = 0; nt < 2; ++nt) {
      const int nidx = wn + nt * 16 + row16;
      #pragma unroll
      for (int r = 0; r < 8; ++r) {
        const int midx = wm + mt * 16 + r + 8 * half;  // C layout: M = r + 8*half
        const float d = xs[midx] + ys[nidx] - 2.0f * acc[mt][nt].f[r];
        lsum += __expf(-0.5f * d);
      }
    }
  lsum *= w;

  __syncthreads();
  red[t] = lsum;
  __syncthreads();
  #pragma unroll
  for (int s = NT / 2; s > 0; s >>= 1) {
    if (t < s) red[t] += red[t + s];
    __syncthreads();
  }
  if (t == 0) partials[pidx] = red[0];
}

// ---------------------------------------------------------------------------
// Kernel 3: deterministic fixed-order reduction of block partials -> mean.
// ---------------------------------------------------------------------------
__global__ __launch_bounds__(256) void mmd_reduce(
    const float* __restrict__ partials, int n, float* __restrict__ out) {
  __shared__ float red[256];
  const int t = threadIdx.x;
  float s = 0.0f;
  for (int i = t; i < n; i += 256) s += partials[i];
  red[t] = s;
  __syncthreads();
  #pragma unroll
  for (int k = 128; k > 0; k >>= 1) {
    if (t < k) red[t] += red[t + k];
    __syncthreads();
  }
  if (t == 0) out[0] = red[0] * (1.0f / ((float)N * (float)N));
}

// ---------------------------------------------------------------------------
extern "C" void kernel_launch(void* const* d_in, const int* in_sizes, int n_in,
                              void* d_out, int out_size, void* d_ws, size_t ws_size,
                              hipStream_t stream) {
  (void)in_sizes; (void)n_in; (void)out_size; (void)ws_size;
  const float* fs = (const float*)d_in[0];
  const float* ft = (const float*)d_in[1];

  // Workspace layout (~16.2 MB): bf16 hi/lo copies, row norms, partials.
  const size_t nd = (size_t)N * D;
  unsigned short* s_hi = (unsigned short*)d_ws;
  unsigned short* s_lo = s_hi + nd;
  unsigned short* t_hi = s_lo + nd;
  unsigned short* t_lo = t_hi + nd;
  float* x2s      = (float*)(t_lo + nd);
  float* x2t      = x2s + N;
  float* partials = x2t + N;

  const int tiles = N / BT;                 // 64
  const int nPart = 3 * tiles * tiles;      // 12288

  mmd_prep<<<dim3(N, 2), D, 0, stream>>>(fs, ft, s_hi, s_lo, t_hi, t_lo, x2s, x2t);
  mmd_tiles<<<dim3(tiles, tiles, 3), NT, 0, stream>>>(
      s_hi, s_lo, t_hi, t_lo, x2s, x2t, partials);
  mmd_reduce<<<1, 256, 0, stream>>>(partials, nPart, (float*)d_out);
}
